// SuperONN2d_30185030157073
// MI455X (gfx1250) — matrix operand order repaired and checked by reading, not compiled, not run on hardware
//
#include <hip/hip_runtime.h>

// SuperONN2d fused: bilinear-shift + Maclaurin powers -> implicit-GEMM 3x3 conv
// via V_WMMA_F32_16X16X32_F16 (f16 in / f32 accumulate) on gfx1250 (wave32).
// Tile staging uses GLOBAL_LOAD_ASYNC_TO_LDS_B128 (ASYNCcnt) + double-buffered
// LDS so the memory pipe overlaps the WMMA stream.
//
// Roofline: conv = 88.7 GFLOP; f16 dense WMMA ~5 PFLOPS -> ~18 us compute,
// ~230 MB HBM traffic -> ~10 us; compute-bound -> WMMA f16 w/ f32 acc.

typedef __attribute__((ext_vector_type(16))) _Float16 v16h;
typedef __attribute__((ext_vector_type(8)))  _Float16 v8h;
typedef __attribute__((ext_vector_type(8)))  float    v8f;

#define NB    16
#define CIN   64
#define CK    192            // Q*CIN (powers concat)
#define HH    112
#define WW    112
#define HP    114            // padded (halo 1)
#define WP    114
#define COUT  128
#define KTOT  1728           // 9 taps * 192
#define NPOS  (NB*HH*WW)     // 200704 = 1568 * 128
#define NCHUNK 54            // KTOT / 32

#define Y_POS_STRIDE CK
#define Y_ROW_STRIDE (WP*CK)
#define Y_IMG_STRIDE (HP*WP*CK)

#define LDS_STRIDE 48                 // 32 data halves + 16 pad; 96B rows
#define ABUF_HALVES (COUT * LDS_STRIDE)
#define BBUF_HALVES (128  * LDS_STRIDE)
#define ABUF_BYTES  (ABUF_HALVES * 2)
#define BBUF_BYTES  (BBUF_HALVES * 2)

// ---- gfx1250 async global->LDS copy (16B per lane), tracked by ASYNCcnt ----
__device__ __forceinline__ void async_copy16(unsigned ldsAddr, const void* g) {
    asm volatile("global_load_async_to_lds_b128 %0, %1, off"
                 :: "v"(ldsAddr), "v"((unsigned long long)(size_t)g)
                 : "memory");
}
__device__ __forceinline__ void wait_async_le4() {
    asm volatile("s_wait_asynccnt 0x4" ::: "memory");
}
__device__ __forceinline__ void wait_async_le0() {
    asm volatile("s_wait_asynccnt 0x0" ::: "memory");
}

// ---------------- stage 1: bilinear shift + powers -> padded NHWC f16 -------
__device__ __forceinline__ float sampleX(const float* __restrict__ xc, int yi, int xi) {
    bool ok = (yi >= 0) & (yi < HH) & (xi >= 0) & (xi < WW);
    int yc  = min(max(yi, 0), HH - 1);
    int xcl = min(max(xi, 0), WW - 1);
    float v = xc[yc * WW + xcl];
    return ok ? v : 0.0f;
}

__global__ __launch_bounds__(256)
void stage1_shift_pow(const float* __restrict__ x,
                      const float* __restrict__ shifts,
                      _Float16* __restrict__ y) {
    __shared__ float sShift[2 * CIN];
    int t = threadIdx.x;
    if (t < 2 * CIN) sShift[t] = shifts[t] * 4.0f;   // MAX_SHIFT = 4
    __syncthreads();

    int gid = blockIdx.x * 256 + t;
    if (gid >= NB * HP * WP) return;
    int n  = gid / (HP * WP);
    int r  = gid % (HP * WP);
    int hp = r / WP, wp = r % WP;
    _Float16* dst = y + (size_t)gid * CK;

    if (hp == 0 || hp == HP - 1 || wp == 0 || wp == WP - 1) {
        uint4 z = make_uint4(0u, 0u, 0u, 0u);
        uint4* p = (uint4*)dst;
        #pragma unroll
        for (int i = 0; i < CK / 8; ++i) p[i] = z;   // zero halo
        return;
    }
    int h = hp - 1, w = wp - 1;
    const float* xn = x + (size_t)n * CIN * HH * WW;

    for (int cg = 0; cg < CIN; cg += 8) {
        v8h p1, p2, p3;
        #pragma unroll
        for (int u = 0; u < 8; ++u) {
            int ci = cg + u;
            float sx = sShift[2 * ci + 0], sy = sShift[2 * ci + 1];
            float jx = (float)w + sx, jy = (float)h + sy;
            float x0f = floorf(jx), y0f = floorf(jy);
            float wx = jx - x0f,  wy = jy - y0f;
            int x0 = (int)x0f, y0 = (int)y0f;
            const float* xc = xn + (size_t)ci * (HH * WW);
            float g00 = sampleX(xc, y0,     x0);
            float g01 = sampleX(xc, y0,     x0 + 1);
            float g10 = sampleX(xc, y0 + 1, x0);
            float g11 = sampleX(xc, y0 + 1, x0 + 1);
            float v  = (1.f - wy) * ((1.f - wx) * g00 + wx * g01)
                     +        wy  * ((1.f - wx) * g10 + wx * g11);
            float v2 = v * v;
            p1[u] = (_Float16)v;
            p2[u] = (_Float16)v2;
            p3[u] = (_Float16)(v2 * v);
        }
        *(v8h*)(dst + cg)            = p1;
        *(v8h*)(dst + CIN + cg)      = p2;
        *(v8h*)(dst + 2 * CIN + cg)  = p3;
    }
}

// ---------------- weights: f32 OIHW -> f16 [co][tap*192+ci] -----------------
__global__ __launch_bounds__(256)
void prep_weights(const float* __restrict__ wsrc, _Float16* __restrict__ wh) {
    int e = blockIdx.x * 256 + threadIdx.x;
    if (e >= COUT * KTOT) return;
    int co = e / KTOT, kk = e % KTOT;
    int tap = kk / CK, ci = kk % CK;
    int kh = tap / 3, kw = tap % 3;
    wh[e] = (_Float16)wsrc[((co * CK + ci) * 3 + kh) * 3 + kw];
}

// ---------------- implicit-GEMM conv via WMMA -------------------------------
// LDS A rows are stored k-linear so each lane's 16 A halves are one
// contiguous 32B region: lane<16 reads K0..15, lane>=16 reads K16..31.
// LDS B rows stay k-linear: lane<16 reads K0..15, lane>=16 reads K16..31.
// All fragment reads are a single 32B-aligned v16h load.
__global__ __launch_bounds__(256)
void conv_wmma(const _Float16* __restrict__ y, const _Float16* __restrict__ wh,
               const float* __restrict__ bias, float* __restrict__ out) {
    __shared__ __align__(32) _Float16 ldsA[2 * ABUF_HALVES]; // 2x 128co x 32k
    __shared__ __align__(32) _Float16 ldsB[2 * BBUF_HALVES]; // 2x 128pos x 32k
    __shared__ float sBias[COUT];

    int t = threadIdx.x;
    if (t < COUT) sBias[t] = bias[t];

    int lane  = t & 31;
    int wave  = t >> 5;          // 8 waves
    int waveM = wave >> 1;       // 0..3 -> co base 32*waveM
    int waveN = wave & 1;        // 0..1 -> pos base 64*waveN

    // loader coords: each thread moves 16 halves of A and 16 of B per chunk
    int rowId  = t >> 1;         // co row for A / spatial row for B
    int kcHalf = (t & 1) * 16;
    int aSt0 = (t & 1) ? 16 : 0;     // k-linear (see comment)
    int aSt1 = (t & 1) ? 24 : 8;

    int pLoad = blockIdx.x * 128 + rowId;
    int nL = pLoad / (HH * WW);
    int rL = pLoad % (HH * WW);
    int hL = rL / WW, wL = rL % WW;
    size_t bBase = (size_t)nL * Y_IMG_STRIDE + (size_t)hL * Y_ROW_STRIDE
                 + (size_t)wL * Y_POS_STRIDE;         // tap (0,0) in padded y
    const _Float16* aBase = wh + (size_t)rowId * KTOT + kcHalf;

    // LDS byte addresses for this thread's async destinations
    unsigned ldsABase = (unsigned)(size_t)(void*)ldsA;
    unsigned ldsBBase = (unsigned)(size_t)(void*)ldsB;
    unsigned aOff0 = (unsigned)((rowId * LDS_STRIDE + aSt0) * 2);
    unsigned aOff1 = (unsigned)((rowId * LDS_STRIDE + aSt1) * 2);
    unsigned bOff0 = (unsigned)((rowId * LDS_STRIDE + kcHalf) * 2);

    v8f acc[2][4];
    v8f zf = {0.f, 0.f, 0.f, 0.f, 0.f, 0.f, 0.f, 0.f};
    #pragma unroll
    for (int i = 0; i < 2; ++i)
        #pragma unroll
        for (int j = 0; j < 4; ++j) acc[i][j] = zf;

    // per-lane fragment half-offsets (constant across chunks)
    int aFragOff = ((waveM * 32 + (lane & 15)) * LDS_STRIDE) + ((lane < 16) ? 0 : 16);
    int bFragOff = ((waveN * 64 + (lane & 15)) * LDS_STRIDE) + ((lane < 16) ? 0 : 16);

    // issue async tile loads for chunk c into buffer (c&1): 4 ops/thread
    auto issue = [&](int c) {
        unsigned buf = (unsigned)(c & 1);
        const _Float16* ga = aBase + c * 32;
        async_copy16(ldsABase + buf * ABUF_BYTES + aOff0, ga);
        async_copy16(ldsABase + buf * ABUF_BYTES + aOff1, ga + 8);
        int tap = c / 6, cib = (c % 6) * 32;
        int dh = tap / 3, dw = tap % 3;
        const _Float16* gb = y + bBase + (size_t)(dh * WP + dw) * CK + cib + kcHalf;
        async_copy16(ldsBBase + buf * BBUF_BYTES + bOff0, gb);
        async_copy16(ldsBBase + buf * BBUF_BYTES + bOff0 + 16, gb + 8);
    };

    issue(0);
    for (int c = 0; c < NCHUNK; ++c) {
        if (c + 1 < NCHUNK) {
            issue(c + 1);          // prefetch next chunk into other buffer
            wait_async_le4();      // chunk c's 4 ops landed (in-order), next 4 fly
        } else {
            wait_async_le0();
        }
        __syncthreads();           // all waves' chunk-c data visible

        int cb = (c & 1);
        const _Float16* A = ldsA + cb * ABUF_HALVES;
        const _Float16* B = ldsB + cb * BBUF_HALVES;
        v16h afr0 = *(const v16h*)&A[aFragOff];
        v16h afr1 = *(const v16h*)&A[aFragOff + 16 * LDS_STRIDE];
        #pragma unroll
        for (int ni = 0; ni < 4; ++ni) {
            v16h bfr = *(const v16h*)&B[bFragOff + ni * 16 * LDS_STRIDE];
            acc[0][ni] = __builtin_amdgcn_wmma_f32_16x16x32_f16(
                false, afr0, false, bfr, (short)0, acc[0][ni], false, false);
            acc[1][ni] = __builtin_amdgcn_wmma_f32_16x16x32_f16(
                false, afr1, false, bfr, (short)0, acc[1][ni], false, false);
        }
        __syncthreads();           // done reading buf before it is re-targeted
    }

    // epilogue: D layout (ISA): lane 0..15 -> N=lane, M=vgpr; lane>=16 -> M=8+vgpr
    int hiHalf = lane >> 4;
    #pragma unroll
    for (int mi = 0; mi < 2; ++mi) {
        #pragma unroll
        for (int ni = 0; ni < 4; ++ni) {
            int jcol = waveN * 64 + ni * 16 + (lane & 15);
            int p = blockIdx.x * 128 + jcol;
            int n = p / (HH * WW);
            int r = p % (HH * WW);
            int h = r / WW, w = r % WW;
            #pragma unroll
            for (int q = 0; q < 8; ++q) {
                int co = waveM * 32 + mi * 16 + hiHalf * 8 + q;
                out[(((size_t)n * COUT + co) * HH + h) * WW + w] =
                    acc[mi][ni][q] + sBias[co];
            }
        }
    }
}

// ---------------- launch ----------------------------------------------------
extern "C" void kernel_launch(void* const* d_in, const int* in_sizes, int n_in,
                              void* d_out, int out_size, void* d_ws, size_t ws_size,
                              hipStream_t stream) {
    const float* x      = (const float*)d_in[0];   // (16,64,112,112)
    const float* wsrc   = (const float*)d_in[1];   // (128,192,3,3)
    const float* bias   = (const float*)d_in[2];   // (128,)
    const float* shifts = (const float*)d_in[3];   // (64,2)
    float* out = (float*)d_out;                    // (16,128,112,112)

    size_t yBytes = (size_t)NB * HP * WP * CK * sizeof(_Float16); // ~79.85 MB
    _Float16* y  = (_Float16*)d_ws;
    _Float16* wh = (_Float16*)((char*)d_ws + ((yBytes + 255) & ~(size_t)255));

    int nPad = NB * HP * WP;
    stage1_shift_pow<<<(nPad + 255) / 256, 256, 0, stream>>>(x, shifts, y);
    prep_weights<<<(COUT * KTOT + 255) / 256, 256, 0, stream>>>(wsrc, wh);
    conv_wmma<<<NPOS / 128, 256, 0, stream>>>(y, wh, bias, out);
}
